// LorentzMultiheadAttention_56788057588123
// MI455X (gfx1250) — matrix operand order repaired and checked by reading, not compiled, not run on hardware
//
#include <hip/hip_runtime.h>
#include <hip/hip_bf16.h>
#include <math.h>

// ---- problem constants (match reference) ----
constexpr int kB = 2;
constexpr int kN = 2048;
constexpr int kH = 8;
constexpr int kD = 64;        // per-head dim incl. time
constexpr int kE = 512;       // embed dim
constexpr int kDO = kH * (kD - 1);  // 504 projection outputs
constexpr int kDOpad = 512;         // padded to multiple of 16

typedef __attribute__((ext_vector_type(16))) _Float16 v16h;
typedef __attribute__((ext_vector_type(8)))  _Float16 v8h;
typedef __attribute__((ext_vector_type(8)))  float    v8f;

static __device__ __forceinline__ v16h cat8(v8h a, v8h b) {
  return __builtin_shufflevector(a, b, 0,1,2,3,4,5,6,7,8,9,10,11,12,13,14,15);
}

static __device__ __forceinline__ v8f wmma_f16(v16h a, v16h b, v8f c) {
  // D = A(16x32 f16) * B(32x16 f16) + C(16x16 f32)
  return __builtin_amdgcn_wmma_f32_16x16x32_f16(
      /*neg_a=*/false, a, /*neg_b=*/false, b,
      /*c_mod=*/(short)0, c, /*reuse_a=*/false, /*reuse_b=*/false);
}

// =====================================================================
// Kernel 0: convert Wq/Wk/Wv (f32 [504][512]) -> zero-padded f16
//           Wh[3][512][512]; row-major = B-fragment friendly for X*W^T
// =====================================================================
__global__ __launch_bounds__(256) void wconv_kernel(
    const float* __restrict__ Wq, const float* __restrict__ Wk,
    const float* __restrict__ Wv, _Float16* __restrict__ Wh)
{
  const int idx = blockIdx.x * 256 + threadIdx.x;   // over 3*512*512
  if (idx >= 3 * kDOpad * kE) return;
  const int m    = idx / (kDOpad * kE);
  const int rest = idx - m * (kDOpad * kE);
  const int n    = rest >> 9;         // padded out-channel 0..511
  const int k    = rest & (kE - 1);
  const float* W = (m == 0) ? Wq : (m == 1) ? Wk : Wv;
  const float v  = (n < kDO) ? W[(size_t)n * kE + k] : 0.0f;
  Wh[idx] = (_Float16)v;
}

// =====================================================================
// Kernel 1: WMMA projection GEMM + hyperboloid lift, fused.
//   grid = (M/16 row tiles, 3 matrices), block = 128 (4 waves).
//   Each wave owns 8 column tiles; A-fragment converted f32->f16 on load.
//   Outputs:
//     Qh[b][h][n][64] f16, dim0 = -sqrt(||q||^2+1)   (Lorentz sign folded)
//     Kh[b][h][n][64] f16, dim0 = +sqrt(||k||^2+1)
//     Vt[b][h][64][n] f16 (dim-major for contiguous PV B-fragments)
// =====================================================================
__global__ __launch_bounds__(128) void proj_wmma_kernel(
    const float* __restrict__ qin, const float* __restrict__ sin_,
    const _Float16* __restrict__ Wh,
    const float* __restrict__ bq, const float* __restrict__ bk,
    const float* __restrict__ bv,
    _Float16* __restrict__ Qh, _Float16* __restrict__ Kh,
    _Float16* __restrict__ Vt)
{
  const int m    = blockIdx.y;          // 0=q, 1=k, 2=v
  const int r0   = blockIdx.x * 16;     // row-tile base (flat b*kN+n)
  const int tid  = threadIdx.x;
  const int w    = tid >> 5;            // wave id 0..3
  const int lane = tid & 31;
  const int col  = lane & 15;
  const int hi   = lane >> 4;

  __shared__ alignas(16) float T[16][kDOpad];   // 32 KB result tile

  const float* X      = (m == 0) ? qin : sin_;
  const float* Xrow   = X + (size_t)(r0 + col) * kE;   // A-frag row M = col
  const _Float16* Wm  = Wh + (size_t)m * kDOpad * kE;

  v8f acc[8];
#pragma unroll
  for (int t = 0; t < 8; ++t) acc[t] = (v8f){};

  for (int ks = 0; ks < kE / 32; ++ks) {        // 16 K-steps of 32
    // A-fragment: two contiguous 8-float runs, converted to f16
    const float* ap = Xrow + ks * 32 + 16 * hi;
    const v8h alo = __builtin_convertvector(*(const v8f*)ap,        v8h);
    const v8h ahi = __builtin_convertvector(*(const v8f*)(ap + 8), v8h);
    const v16h a  = cat8(alo, ahi);
#pragma unroll
    for (int t = 0; t < 8; ++t) {
      const int ct = w + 4 * t;                 // this wave's column tile
      const _Float16* bp =
          Wm + (size_t)(ct * 16 + col) * kE + ks * 32 + 16 * hi;
      acc[t] = wmma_f16(a, *(const v16h*)bp, acc[t]);
    }
  }

#pragma unroll
  for (int t = 0; t < 8; ++t) {
#pragma unroll
    for (int r = 0; r < 8; ++r)
      T[r + 8 * hi][(w + 4 * t) * 16 + col] = acc[t][r];
  }
  __syncthreads();

  // ---- lift epilogue: one thread per (row, head) ----
  const float* bb = (m == 0) ? bq : (m == 1) ? bk : bv;
  const int row = tid >> 3;            // 0..15
  const int h   = tid & 7;
  const int g   = r0 + row;            // flat b*kN + n
  const int b   = g >> 11;             // kN = 2048
  const int n   = g & (kN - 1);
  const size_t bhn  = (size_t)(b * kH + h) * kN + n;
  const size_t vrow = (size_t)(b * kH + h) * 64;

  float ssum = 0.0f;
  if (m == 0) {
    for (int j = 0; j < kD - 1; ++j) {
      const float val = T[row][h * (kD - 1) + j] + bb[h * (kD - 1) + j];
      ssum = fmaf(val, val, ssum);
      Qh[bhn * 64 + 1 + j] = (_Float16)val;
    }
    Qh[bhn * 64] = (_Float16)(-sqrtf(ssum + 1.0f));   // -time (sign folded)
  } else if (m == 1) {
    for (int j = 0; j < kD - 1; ++j) {
      const float val = T[row][h * (kD - 1) + j] + bb[h * (kD - 1) + j];
      ssum = fmaf(val, val, ssum);
      Kh[bhn * 64 + 1 + j] = (_Float16)val;
    }
    Kh[bhn * 64] = (_Float16)sqrtf(ssum + 1.0f);
  } else {
    for (int j = 0; j < kD - 1; ++j) {
      const float val = T[row][h * (kD - 1) + j] + bb[h * (kD - 1) + j];
      ssum = fmaf(val, val, ssum);
      Vt[(vrow + 1 + j) * kN + n] = (_Float16)val;
    }
    Vt[vrow * kN + n] = (_Float16)sqrtf(ssum + 1.0f);
  }
}

// =====================================================================
// Kernel 2: flash attention with WMMA + online softmax + per-head centroid
//   one wave (32 threads) per 16-query tile of one (b,h)
// =====================================================================
__global__ __launch_bounds__(32) void attn_kernel(
    const _Float16* __restrict__ Qh, const _Float16* __restrict__ Kh,
    const _Float16* __restrict__ Vt, float* __restrict__ Hout,
    const float* __restrict__ scale_p, const float* __restrict__ bias_p)
{
  const int lane = threadIdx.x;
  const int col  = lane & 15;          // A-frag row M / B-frag column N
  const int hi   = lane >> 4;          // lane-half selector
  const int bh   = blockIdx.y;         // b*kH + h
  const int b    = bh >> 3;            // kH = 8
  const int h    = bh & 7;
  const int q0   = blockIdx.x * 16;

  const float inv_scale = 1.0f / scale_p[0];
  const float sc  = 2.0f * inv_scale;            // att = (2 + 2*s)/scale + bias
  const float off = sc + bias_p[0];

  __shared__ alignas(32) float    S_lds[16][32];
  __shared__ alignas(32) _Float16 P_lds[16][32];
  __shared__ float c_lds[16];
  __shared__ float l_lds[16];
  __shared__ float s_lds[16];
  __shared__ alignas(16) float Ot[16][64];

  // ---- Q A-fragments (16x32 f16 layout): lane holds rows of chunk pairs ----
  const _Float16* qrow = Qh + ((size_t)bh * kN + q0 + col) * 64;
  const v16h aq0 = cat8(*(const v8h*)(qrow +      16 * hi),
                        *(const v8h*)(qrow + 8 + 16 * hi));   // dims 0..31
  const v16h aq1 = cat8(*(const v8h*)(qrow + 32 + 16 * hi),
                        *(const v8h*)(qrow + 40 + 16 * hi));   // dims 32..63

  const _Float16* kbase = Kh + (size_t)bh * kN * 64;
  const _Float16* vbase = Vt + (size_t)bh * 64 * kN;

  float m_run = -INFINITY, l_run = 0.0f;
  v8f o0 = {}, o1 = {}, o2 = {}, o3 = {};   // 16x64 fp32 accumulator

  for (int j0 = 0; j0 < kN; j0 += 32) {
    // prefetch next key block (global_prefetch_b8 path)
    if (j0 + 32 < kN) {
      __builtin_prefetch(kbase + (size_t)(j0 + 32 + col) * 64, 0, 1);
      __builtin_prefetch(vbase + (size_t)col * kN + j0 + 32, 0, 1);
    }

    // ---- scores: S(16x32) = Q' * K^T over 64 dims ----
    const _Float16* krowL = kbase + (size_t)(j0 + col) * 64;
    const _Float16* krowR = krowL + 16 * 64;
    const v16h bL0 = *(const v16h*)(krowL +      16 * hi);
    const v16h bL1 = *(const v16h*)(krowL + 32 + 16 * hi);
    const v16h bR0 = *(const v16h*)(krowR +      16 * hi);
    const v16h bR1 = *(const v16h*)(krowR + 32 + 16 * hi);

    v8f sL = {};
    sL = wmma_f16(aq0, bL0, sL);
    sL = wmma_f16(aq1, bL1, sL);
    v8f sR = {};
    sR = wmma_f16(aq0, bR0, sR);
    sR = wmma_f16(aq1, bR1, sR);

#pragma unroll
    for (int r = 0; r < 8; ++r) {
      S_lds[r + 8 * hi][col]      = fmaf(sL[r], sc, off);
      S_lds[r + 8 * hi][col + 16] = fmaf(sR[r], sc, off);
    }
    __syncthreads();

    // ---- online softmax: lanes 0..15 each own one query row ----
    if (lane < 16) {
      float rmax = -INFINITY;
#pragma unroll 8
      for (int t = 0; t < 32; ++t) rmax = fmaxf(rmax, S_lds[lane][t]);
      const float mnew = fmaxf(m_run, rmax);
      const float corr = __expf(m_run - mnew);   // 0 on first iter (-inf)
      float psum = 0.0f;
#pragma unroll 8
      for (int t = 0; t < 32; ++t) {
        const float p = __expf(S_lds[lane][t] - mnew);
        P_lds[lane][t] = (_Float16)p;
        psum += p;
      }
      l_run = l_run * corr + psum;
      m_run = mnew;
      c_lds[lane] = corr;
    }
    __syncthreads();

    // rescale accumulator by per-row correction
#pragma unroll
    for (int r = 0; r < 8; ++r) {
      const float cr = c_lds[r + 8 * hi];
      o0[r] *= cr; o1[r] *= cr; o2[r] *= cr; o3[r] *= cr;
    }

    // P A-fragment (16x32 f16) from LDS
    const v16h ap = cat8(*(const v8h*)&P_lds[col][    16 * hi],
                         *(const v8h*)&P_lds[col][8 + 16 * hi]);

    // V B-fragments: contiguous key-runs thanks to dim-major Vt layout
    const _Float16* vcol = vbase + (size_t)col * kN + j0 + 16 * hi;
    const v16h bv0 = *(const v16h*)(vcol);
    const v16h bv1 = *(const v16h*)(vcol + 16 * kN);
    const v16h bv2 = *(const v16h*)(vcol + 32 * kN);
    const v16h bv3 = *(const v16h*)(vcol + 48 * kN);

    o0 = wmma_f16(ap, bv0, o0);
    o1 = wmma_f16(ap, bv1, o1);
    o2 = wmma_f16(ap, bv2, o2);
    o3 = wmma_f16(ap, bv3, o3);
  }

  // ---- finalize: divide by softmax denominator, per-head centroid ----
  if (lane < 16) l_lds[lane] = l_run;
  __syncthreads();
#pragma unroll
  for (int r = 0; r < 8; ++r) {
    const float li = 1.0f / l_lds[r + 8 * hi];
    Ot[r + 8 * hi][ 0 + col] = o0[r] * li;
    Ot[r + 8 * hi][16 + col] = o1[r] * li;
    Ot[r + 8 * hi][32 + col] = o2[r] * li;
    Ot[r + 8 * hi][48 + col] = o3[r] * li;
  }
  __syncthreads();

  if (lane < 16) {
    float ssum = 0.0f;
#pragma unroll 8
    for (int d = 0; d < 64; ++d) { const float x = Ot[lane][d]; ssum = fmaf(x, x, ssum); }
    const float a0 = Ot[lane][0];
    const float inner = ssum - 2.0f * a0 * a0;   // Lorentzian <ave,ave>
    s_lds[lane] = 1.0f / sqrtf(fmaxf(fabsf(inner), 1e-8f));  // sqrt(C)=1
  }
  __syncthreads();

  for (int idx = lane; idx < 16 * 64; idx += 32) {
    const int mm = idx >> 6, d = idx & 63;
    Hout[(((size_t)b * kN + q0 + mm) * kH + h) * 64 + d] = Ot[mm][d] * s_lds[mm];
  }
}

// =====================================================================
// Kernel 3: mean over heads + second Lorentz centroid -> d_out [B,N,64]
// =====================================================================
__global__ __launch_bounds__(64) void head_reduce_kernel(
    const float* __restrict__ Hout, float* __restrict__ out)
{
  const int d  = threadIdx.x;
  const int bn = blockIdx.x;
  __shared__ float av[64];
  __shared__ float red[64];

  const float* base = Hout + (size_t)bn * kH * 64 + d;
  float a = 0.0f;
#pragma unroll
  for (int hh = 0; hh < kH; ++hh) a += base[hh * 64];
  a *= (1.0f / kH);

  av[d]  = a;
  red[d] = a * a;
  __syncthreads();
#pragma unroll
  for (int s = 32; s > 0; s >>= 1) {
    if (d < s) red[d] += red[d + s];
    __syncthreads();
  }
  const float inner = red[0] - 2.0f * av[0] * av[0];
  const float den   = sqrtf(fmaxf(fabsf(inner), 1e-8f));
  out[(size_t)bn * 64 + d] = av[d] / den;
}

// =====================================================================
extern "C" void kernel_launch(void* const* d_in, const int* in_sizes, int n_in,
                              void* d_out, int out_size, void* d_ws, size_t ws_size,
                              hipStream_t stream) {
  const float* qin   = (const float*)d_in[0];
  const float* sin_  = (const float*)d_in[1];
  const float* Wq    = (const float*)d_in[2];
  const float* bq    = (const float*)d_in[3];
  const float* Wk    = (const float*)d_in[4];
  const float* bk    = (const float*)d_in[5];
  const float* Wv    = (const float*)d_in[6];
  const float* bv    = (const float*)d_in[7];
  const float* scale = (const float*)d_in[8];
  const float* bias  = (const float*)d_in[9];

  // workspace layout: Qh(4MB) | Kh(4MB) | Vt(4MB) | Hout(8MB) | Wh(1.5MB)
  char* ws = (char*)d_ws;
  _Float16* Qh = (_Float16*)ws;
  _Float16* Kh = Qh + (size_t)kB * kH * kN * 64;
  _Float16* Vt = Kh + (size_t)kB * kH * kN * 64;
  float*  Hout = (float*)(Vt + (size_t)kB * kH * 64 * kN);
  _Float16* Wh = (_Float16*)(Hout + (size_t)kB * kN * kH * 64);

  wconv_kernel<<<(3 * kDOpad * kE + 255) / 256, 256, 0, stream>>>(
      Wq, Wk, Wv, Wh);

  proj_wmma_kernel<<<dim3(kB * kN / 16, 3), 128, 0, stream>>>(
      qin, sin_, Wh, bq, bk, bv, Qh, Kh, Vt);

  attn_kernel<<<dim3(kN / 16, kB * kH), 32, 0, stream>>>(
      Qh, Kh, Vt, Hout, scale, bias);

  head_reduce_kernel<<<kB * kN, 64, 0, stream>>>(
      Hout, (float*)d_out);
}
